// PointCluOT_32710470926932
// MI455X (gfx1250) — matrix operand order repaired and checked by reading, not compiled, not run on hardware
//
#include <hip/hip_runtime.h>

static constexpr int B_ = 8, D_ = 1024, N_ = 4096, K_ = 64, H_ = 512;

typedef __attribute__((ext_vector_type(16))) __bf16 v16bf;
typedef __attribute__((ext_vector_type(8)))  float v8f;

__device__ __forceinline__ unsigned short f2bf(float f) {
  union { float f; unsigned u; } x; x.f = f;
  unsigned r = x.u + 0x7FFFu + ((x.u >> 16) & 1u);
  return (unsigned short)(r >> 16);
}

__device__ __forceinline__ unsigned pk2bf(float lo, float hi) {
  return (unsigned)f2bf(lo) | ((unsigned)f2bf(hi) << 16);
}

union Frag16 { v16bf v; unsigned u[8]; };

// ---------------------------------------------------------------------------
// Generic stride-parameterized bf16 WMMA GEMM: C[m,n] = sum_k A[m,k]*B[k,n]+bias[m]
// A elem at A[m*aRs + k*aCs], B elem at B[k*bRs + n*bCs], C at C[m*cRs + n*cCs].
// Block tile 64x64, 256 threads (8 wave32 waves), each wave: 16x32 via 2 WMMAs.
// Double-buffered LDS staging (f32 -> packed bf16 dwords), 1 barrier per chunk.
// ---------------------------------------------------------------------------
__global__ __launch_bounds__(256) void gemm_bf16_wmma(
    const float* __restrict__ A, long aRs, long aCs, long aBs,
    const float* __restrict__ B, long bRs, long bCs, long bBs,
    float* __restrict__ C, long cRs, long cCs, long cBs,
    const float* __restrict__ bias, int Kd)
{
  __shared__ unsigned short As[2][64][34];   // [buf][m][k], 68B rows keep dword align
  __shared__ unsigned short Bs[2][64][34];   // [buf][n][k]
  const float* Ab = A + (long)blockIdx.z * aBs;
  const float* Bb = B + (long)blockIdx.z * bBs;
  float*       Cb = C + (long)blockIdx.z * cBs;
  const int m0 = blockIdx.y * 64, n0 = blockIdx.x * 64;
  const int tid = threadIdx.x;
  const int w = tid >> 5, lane = tid & 31;
  const int mSub = (w & 3) << 4, nSub = (w >> 2) << 5;
  const int r16 = lane & 15, half = lane >> 4;

  v8f acc0 = {0.f,0.f,0.f,0.f,0.f,0.f,0.f,0.f};
  v8f acc1 = {0.f,0.f,0.f,0.f,0.f,0.f,0.f,0.f};

  // stage one 64x32 A tile + 32x64 B tile (as bf16 pairs -> 32-bit DS stores)
  auto stage = [&](int buf, int k0) {
#pragma unroll
    for (int i = 0; i < 4; ++i) {            // A: 1024 k-pairs
      int p = tid + i * 256;
      int kl = (p & 15) << 1, ml = p >> 4;
      const float* s = Ab + (long)(m0 + ml) * aRs + (long)(k0 + kl) * aCs;
      *(unsigned*)&As[buf][ml][kl] = pk2bf(s[0], s[aCs]);
    }
#pragma unroll
    for (int i = 0; i < 4; ++i) {            // B: 1024 k-pairs, stored [n][k]
      int p = tid + i * 256;
      int kl = (p & 15) << 1, nl = p >> 4;
      const float* s = Bb + (long)(k0 + kl) * bRs + (long)(n0 + nl) * bCs;
      *(unsigned*)&Bs[buf][nl][kl] = pk2bf(s[0], s[bRs]);
    }
  };

  stage(0, 0);
  __syncthreads();

  const int nChunks = Kd >> 5;
  for (int c = 0; c < nChunks; ++c) {
    const int cur = c & 1;
    if (c + 2 < nChunks) {                   // global_prefetch_b8, one buffer ahead
      __builtin_prefetch(Ab + (long)m0 * aRs + (long)((c + 2) << 5) * aCs, 0, 0);
      __builtin_prefetch(Bb + (long)((c + 2) << 5) * bRs + (long)n0 * bCs, 0, 0);
    }
    if (c + 1 < nChunks) stage(cur ^ 1, (c + 1) << 5);

    Frag16 af, bf0, bf1;
    const int ar = mSub + r16;
#pragma unroll
    for (int g = 0; g < 8; ++g) {
      // A 16x32 bf16 layout: lanes 0-15 M=row; v0-3: K=0..7(+8*half); v4-7: K=16..23(+8*half)
      int kb = (g < 4) ? (g << 1) : (16 + ((g - 4) << 1));
      af.u[g]  = *(const unsigned int*)&As[cur][ar][kb + (half << 3)];
      // B 32x16 bf16 layout: lane = N col; same K order as A: v0-3: K=0..7(+8*half); v4-7: K=16..23(+8*half)
      int k2 = kb + (half << 3);
      bf0.u[g] = *(const unsigned int*)&Bs[cur][nSub + r16][k2];
      bf1.u[g] = *(const unsigned int*)&Bs[cur][nSub + 16 + r16][k2];
    }
    acc0 = __builtin_amdgcn_wmma_f32_16x16x32_bf16(false, af.v, false, bf0.v, (short)0, acc0, false, false);
    acc1 = __builtin_amdgcn_wmma_f32_16x16x32_bf16(false, af.v, false, bf1.v, (short)0, acc1, false, false);
    __syncthreads();                         // orders: reads(cur) < writes(cur) next iter,
  }                                          //         writes(nxt) < reads(nxt) next iter

  // C/D 16x16 f32 layout: N = lane&15, M = r + 8*(lane>>4)
  const int n = n0 + nSub + r16;
#pragma unroll
  for (int r = 0; r < 8; ++r) {
    int m = m0 + mSub + r + (half << 3);
    float bv = bias ? bias[m] : 0.f;
    Cb[(long)m * cRs + (long)n * cCs]        = acc0[r] + bv;
    Cb[(long)m * cRs + (long)(n + 16) * cCs] = acc1[r] + bv;
  }
}

// ---------------------------------------------------------------------------
__global__ void zero_f(float* p, int n) {
  int t = blockIdx.x * 256 + threadIdx.x;
  if (t < n) p[t] = 0.f;
}

__global__ __launch_bounds__(256) void bn_stats_k(const float* __restrict__ x,
                                                  float* mean, float* rstd) {
  int c = blockIdx.x;
  float a = 0.f, b = 0.f;
  for (int i = threadIdx.x; i < B_ * N_; i += 256) {
    int bi = i >> 12, n = i & (N_ - 1);
    float v = x[((long)bi * H_ + c) * N_ + n];
    a += v; b += v * v;
  }
  __shared__ float s1[256], s2[256];
  s1[threadIdx.x] = a; s2[threadIdx.x] = b; __syncthreads();
  for (int st = 128; st > 0; st >>= 1) {
    if (threadIdx.x < st) { s1[threadIdx.x] += s1[threadIdx.x + st]; s2[threadIdx.x] += s2[threadIdx.x + st]; }
    __syncthreads();
  }
  if (threadIdx.x == 0) {
    float inv = 1.f / (float)(B_ * N_);
    float m = s1[0] * inv;
    float v = s2[0] * inv - m * m;
    mean[c] = m; rstd[c] = rsqrtf(v + 1e-5f);
  }
}

__global__ void bn_relu_k(float* x, const float* mean, const float* rstd,
                          const float* g, const float* be) {
  long t = (long)blockIdx.x * 256 + threadIdx.x;
  int c = (int)((t >> 12) & (H_ - 1));
  float v = (x[t] - mean[c]) * rstd[c] * g[c] + be[c];
  x[t] = fmaxf(v, 0.f);
}

__global__ void softmax_k(const float* __restrict__ logits, float* __restrict__ score) {
  int t = blockIdx.x * 256 + threadIdx.x;        // (b,n)
  int b = t >> 12, n = t & (N_ - 1);
  const float* L = logits + (long)b * K_ * N_ + n;
  float mx = -3.4e38f;
  for (int k = 0; k < K_; ++k) mx = fmaxf(mx, L[(long)k * N_]);
  float s = 0.f;
  for (int k = 0; k < K_; ++k) s += expf(L[(long)k * N_] - mx);
  float inv = 1.f / s;
  float* S = score + (long)b * K_ * N_ + n;
  for (int k = 0; k < K_; ++k) S[(long)k * N_] = expf(L[(long)k * N_] - mx) * inv;
}

__global__ __launch_bounds__(256) void pi_k(const float* score, float* pi) {
  int bk = blockIdx.x;
  const float* S = score + (long)bk * N_;
  float a = 0.f;
  for (int n = threadIdx.x; n < N_; n += 256) a += S[n];
  __shared__ float sm[256];
  sm[threadIdx.x] = a; __syncthreads();
  for (int st = 128; st > 0; st >>= 1) {
    if (threadIdx.x < st) sm[threadIdx.x] += sm[threadIdx.x + st];
    __syncthreads();
  }
  if (threadIdx.x == 0) pi[bk] = fmaxf(sm[0], 1e-4f);
}

__global__ __launch_bounds__(256) void mu_xyz_k(const float* score, const float* xyz,
                                                const float* pi, float* muxyz, float* yn) {
  int bk = blockIdx.x; int b = bk >> 6, k = bk & 63;
  const float* S = score + (long)bk * N_;
  const float* X = xyz + (long)b * 3 * N_;
  float a0 = 0.f, a1 = 0.f, a2 = 0.f;
  for (int n = threadIdx.x; n < N_; n += 256) {
    float s = S[n];
    a0 += s * X[n]; a1 += s * X[N_ + n]; a2 += s * X[2 * N_ + n];
  }
  __shared__ float s0[256], s1[256], s2[256];
  s0[threadIdx.x] = a0; s1[threadIdx.x] = a1; s2[threadIdx.x] = a2; __syncthreads();
  for (int st = 128; st > 0; st >>= 1) {
    if (threadIdx.x < st) {
      s0[threadIdx.x] += s0[threadIdx.x + st];
      s1[threadIdx.x] += s1[threadIdx.x + st];
      s2[threadIdx.x] += s2[threadIdx.x + st];
    }
    __syncthreads();
  }
  if (threadIdx.x == 0) {
    float ip = 1.f / pi[bk];
    float m0 = s0[0] * ip, m1 = s1[0] * ip, m2 = s2[0] * ip;
    muxyz[((long)b * 3 + 0) * K_ + k] = m0;
    muxyz[((long)b * 3 + 1) * K_ + k] = m1;
    muxyz[((long)b * 3 + 2) * K_ + k] = m2;
    yn[bk] = m0 * m0 + m1 * m1 + m2 * m2;
  }
}

__global__ void mu_div_k(float* mu, const float* pi) {
  int t = blockIdx.x * 256 + threadIdx.x;        // B*D*K
  int k = t & 63; int b = t >> 16;
  mu[t] *= 1.f / pi[b * K_ + k];
}

__global__ __launch_bounds__(256) void mu_norm_k(const float* mu, float* invnmu) {
  int bk = blockIdx.x; int b = bk >> 6, k = bk & 63;
  const float* Mb = mu + (long)b * D_ * K_ + k;
  float s = 0.f;
  for (int d = threadIdx.x; d < D_; d += 256) { float x = Mb[(long)d * K_]; s += x * x; }
  __shared__ float sm[256];
  sm[threadIdx.x] = s; __syncthreads();
  for (int st = 128; st > 0; st >>= 1) {
    if (threadIdx.x < st) sm[threadIdx.x] += sm[threadIdx.x + st];
    __syncthreads();
  }
  if (threadIdx.x == 0) invnmu[bk] = 1.f / fmaxf(sqrtf(sm[0]), 1e-12f);
}

__global__ void fnorm_k(const float* f, float* invnf) {
  int t = blockIdx.x * 256 + threadIdx.x;        // (b,n)
  int b = t >> 12, n = t & (N_ - 1);
  const float* F = f + (long)b * D_ * N_ + n;
  float s = 0.f;
  for (int d = 0; d < D_; ++d) { float x = F[(long)d * N_]; s += x * x; }
  invnf[t] = 1.f / fmaxf(sqrtf(s), 1e-12f);
}

__global__ void costfe_k(float* cost1, const float* invnf, const float* invnmu) {
  int t = blockIdx.x * 256 + threadIdx.x;        // B*N*K
  int k = t & 63; int bn = t >> 6; int b = bn >> 12;
  cost1[t] = 2.f - 2.f * cost1[t] * invnf[bn] * invnmu[b * K_ + k];
}

__global__ void costeu_k(const float* xyz, const float* muxyz, const float* yn, float* cost0) {
  int t = blockIdx.x * 256 + threadIdx.x;        // (b,n)
  int b = t >> 12, n = t & (N_ - 1);
  const float* X = xyz + (long)b * 3 * N_;
  float x0 = X[n], x1 = X[N_ + n], x2 = X[2 * N_ + n];
  float xn = x0 * x0 + x1 * x1 + x2 * x2;
  float* Cr = cost0 + (long)t * K_;
  const float* M = muxyz + (long)b * 3 * K_;
  for (int k = 0; k < K_; ++k) {
    float d = x0 * M[k] + x1 * M[K_ + k] + x2 * M[2 * K_ + k];
    Cr[k] = xn + yn[b * K_ + k] - 2.f * d;
  }
}

// Sinkhorn on 2 stacked problems: cost[2][B][N][K], u[2][B][N], v[2][B][K]
__global__ void sinku_k(const float* __restrict__ cost, float* __restrict__ u,
                        const float* __restrict__ v) {
  int t = blockIdx.x * 256 + threadIdx.x;        // (s,b,n)
  int s = t >> 15; int b = (t >> 12) & 7;
  const float* Cr = cost + (long)t * K_;
  const float* V = v + (s * B_ + b) * K_;
  float uu = u[t];
  float mx = -3.4e38f;
  for (int m = 0; m < K_; ++m) mx = fmaxf(mx, (uu + V[m] - Cr[m]) * 1000.f);
  float sum = 0.f;
  for (int m = 0; m < K_; ++m) sum += expf((uu + V[m] - Cr[m]) * 1000.f - mx);
  float lse = mx + logf(sum);
  u[t] = 1e-3f * (-8.317766166719343f - lse) + uu;   // log_p = -log(4096)
}

__global__ __launch_bounds__(256) void sinkv_k(const float* __restrict__ cost,
                                               const float* __restrict__ u, float* v) {
  int blk = blockIdx.x;                          // s*512 + b*64 + m
  int s = blk >> 9; int rem = blk & 511; int b = rem >> 6; int m = rem & 63;
  float vv = v[blk];
  const float* Cc = cost + ((long)(s * B_ + b) * N_) * K_ + m;
  const float* U = u + (long)(s * B_ + b) * N_;
  float vals[16];
  float mx = -3.4e38f;
  int c = 0;
  for (int n = threadIdx.x; n < N_; n += 256, ++c) {
    float val = (U[n] + vv - Cc[(long)n * K_]) * 1000.f;
    vals[c] = val; mx = fmaxf(mx, val);
  }
  __shared__ float sm[256];
  sm[threadIdx.x] = mx; __syncthreads();
  for (int st = 128; st > 0; st >>= 1) {
    if (threadIdx.x < st) sm[threadIdx.x] = fmaxf(sm[threadIdx.x], sm[threadIdx.x + st]);
    __syncthreads();
  }
  mx = sm[0]; __syncthreads();
  float sum = 0.f;
  for (int c2 = 0; c2 < 16; ++c2) sum += expf(vals[c2] - mx);
  sm[threadIdx.x] = sum; __syncthreads();
  for (int st = 128; st > 0; st >>= 1) {
    if (threadIdx.x < st) sm[threadIdx.x] += sm[threadIdx.x + st];
    __syncthreads();
  }
  if (threadIdx.x == 0) {
    float lse = mx + logf(sm[0]);
    v[blk] = 1e-3f * (-4.1588830833596715f - lse) + vv;  // log_q = -log(64)
  }
}

__global__ __launch_bounds__(256) void regxyz_k(const float* muxyz, float* acc) {
  int t = blockIdx.x * 256 + threadIdx.x;        // b*K*K
  int b = t >> 12; int m = (t >> 6) & 63; int q = t & 63;
  const float* M = muxyz + (long)b * 3 * K_;
  float g = M[m] * M[q] + M[K_ + m] * M[K_ + q] + M[2 * K_ + m] * M[2 * K_ + q];
  float val = fabsf(g - ((m == q) ? 1.f : 0.f));
  __shared__ float sm[256];
  sm[threadIdx.x] = val; __syncthreads();
  for (int st = 128; st > 0; st >>= 1) {
    if (threadIdx.x < st) sm[threadIdx.x] += sm[threadIdx.x + st];
    __syncthreads();
  }
  if (threadIdx.x == 0) atomicAdd(&acc[2], sm[0] * (1e-7f / 32768.f)); // 0.001*1e-4*mean
}

__global__ __launch_bounds__(256) void regfea_k(const float* mu, const float* invnmu, float* acc) {
  int blk = blockIdx.x;                          // b*64 + m
  int b = blk >> 6; int m = blk & 63;
  int q = threadIdx.x & 63; int part = threadIdx.x >> 6;
  const float* Mb = mu + (long)b * D_ * K_;
  float p = 0.f;
  for (int d = part * 256; d < (part + 1) * 256; ++d)
    p += Mb[(long)d * K_ + m] * Mb[(long)d * K_ + q];
  __shared__ float sm[256];
  sm[threadIdx.x] = p; __syncthreads();
  if (threadIdx.x < 64) {
    float tot = sm[q] + sm[64 + q] + sm[128 + q] + sm[192 + q];
    float g = tot * invnmu[blk] * invnmu[b * K_ + q];
    sm[q] = fabsf(g - ((m == q) ? 1.f : 0.f));
  }
  __syncthreads();
  if (threadIdx.x == 0) {
    float ssum = 0.f;
    for (int i = 0; i < 64; ++i) ssum += sm[i];
    atomicAdd(&acc[3], ssum * (1e-4f / 32768.f));      // 1e-4 * mean over b*K*K
  }
}

__global__ __launch_bounds__(256) void loss_k(const float* logits, const float* cost,
                                              const float* u, const float* v, float* acc) {
  int t = blockIdx.x * 256 + threadIdx.x;        // (b,n)
  int b = t >> 12, n = t & (N_ - 1);
  const float* L = logits + (long)b * K_ * N_ + n;
  float mx = -3.4e38f;
  for (int k = 0; k < K_; ++k) mx = fmaxf(mx, L[(long)k * N_]);
  float s = 0.f;
  for (int k = 0; k < K_; ++k) s += expf(L[(long)k * N_] - mx);
  float lse = mx + logf(s);
  const float* C0 = cost + (long)t * K_;
  const float* C1 = cost + (long)(B_ * N_) * K_ + (long)t * K_;
  float u0 = u[t], u1 = u[B_ * N_ + t];
  const float* V0 = v + b * K_;
  const float* V1 = v + B_ * K_ + b * K_;
  float ax = 0.f, af = 0.f;
  for (int k = 0; k < K_; ++k) {
    float lp = L[(long)k * N_] - lse;
    ax += expf((u0 + V0[k] - C0[k]) * 1000.f) * lp;
    af += expf((u1 + V1[k] - C1[k]) * 1000.f) * lp;
  }
  ax *= (float)N_; af *= (float)N_;
  __shared__ float sx[256], sf[256];
  sx[threadIdx.x] = ax; sf[threadIdx.x] = af; __syncthreads();
  for (int st = 128; st > 0; st >>= 1) {
    if (threadIdx.x < st) { sx[threadIdx.x] += sx[threadIdx.x + st]; sf[threadIdx.x] += sf[threadIdx.x + st]; }
    __syncthreads();
  }
  if (threadIdx.x == 0) { atomicAdd(&acc[0], sx[0]); atomicAdd(&acc[1], sf[0]); }
}

__global__ void fin_k(const float* acc, float* out) {
  out[0] = -(acc[0] + acc[1]) * (1.f / (float)(B_ * N_)) + acc[2] + acc[3];
}

// ---------------------------------------------------------------------------
extern "C" void kernel_launch(void* const* d_in, const int* in_sizes, int n_in,
                              void* d_out, int out_size, void* d_ws, size_t ws_size,
                              hipStream_t stream) {
  (void)in_sizes; (void)n_in; (void)out_size; (void)ws_size;
  const float* feature = (const float*)d_in[0];
  const float* xyz     = (const float*)d_in[1];
  const float* w1 = (const float*)d_in[2];
  const float* b1 = (const float*)d_in[3];
  const float* g1 = (const float*)d_in[4];
  const float* be1 = (const float*)d_in[5];
  const float* w2 = (const float*)d_in[6];
  const float* b2 = (const float*)d_in[7];
  const float* g2 = (const float*)d_in[8];
  const float* be2 = (const float*)d_in[9];
  const float* w3 = (const float*)d_in[10];
  const float* b3 = (const float*)d_in[11];
  float* out = (float*)d_out;

  float* ws = (float*)d_ws;
  long o = 0;
  float* h1     = ws + o; o += (long)B_ * H_ * N_;
  float* h2     = ws + o; o += (long)B_ * H_ * N_;
  float* logits = ws + o; o += (long)B_ * K_ * N_;
  float* score  = ws + o; o += (long)B_ * K_ * N_;
  float* cost   = ws + o; o += 2L * B_ * N_ * K_;   // [0]=eu, [1]=fe (fdot in place)
  float* mu     = ws + o; o += (long)B_ * D_ * K_;
  float* muxyz  = ws + o; o += (long)B_ * 3 * K_;
  float* pi     = ws + o; o += B_ * K_;
  float* yn     = ws + o; o += B_ * K_;
  float* mean1  = ws + o; o += H_;
  float* rstd1  = ws + o; o += H_;
  float* mean2  = ws + o; o += H_;
  float* rstd2  = ws + o; o += H_;
  float* invnf  = ws + o; o += B_ * N_;
  float* invnmu = ws + o; o += B_ * K_;
  float* uvec   = ws + o; o += 2L * B_ * N_;
  float* vvec   = ws + o; o += 2L * B_ * K_;
  float* acc    = ws + o; o += 8;
  float* cost1  = cost + (long)B_ * N_ * K_;

  dim3 blk(256);

  // conv1: h1 = w1[H,D] x feature[D,N]  (per batch)
  gemm_bf16_wmma<<<dim3(N_/64, H_/64, B_), blk, 0, stream>>>(
      w1, (long)D_, 1L, 0L,
      feature, (long)N_, 1L, (long)D_ * N_,
      h1, (long)N_, 1L, (long)H_ * N_, b1, D_);
  bn_stats_k<<<H_, blk, 0, stream>>>(h1, mean1, rstd1);
  bn_relu_k<<<(B_*H_*N_)/256, blk, 0, stream>>>(h1, mean1, rstd1, g1, be1);

  // conv2
  gemm_bf16_wmma<<<dim3(N_/64, H_/64, B_), blk, 0, stream>>>(
      w2, (long)H_, 1L, 0L,
      h1, (long)N_, 1L, (long)H_ * N_,
      h2, (long)N_, 1L, (long)H_ * N_, b2, H_);
  bn_stats_k<<<H_, blk, 0, stream>>>(h2, mean2, rstd2);
  bn_relu_k<<<(B_*H_*N_)/256, blk, 0, stream>>>(h2, mean2, rstd2, g2, be2);

  // conv3 -> logits [B,K,N]
  gemm_bf16_wmma<<<dim3(N_/64, 1, B_), blk, 0, stream>>>(
      w3, (long)H_, 1L, 0L,
      h2, (long)N_, 1L, (long)H_ * N_,
      logits, (long)N_, 1L, (long)K_ * N_, b3, H_);

  softmax_k<<<(B_*N_)/256, blk, 0, stream>>>(logits, score);
  pi_k<<<B_*K_, blk, 0, stream>>>(score, pi);
  mu_xyz_k<<<B_*K_, blk, 0, stream>>>(score, xyz, pi, muxyz, yn);

  // mu_fea[d,k] = sum_n feature[d,n] * score[k,n]   (A=feature, B=score^T via strides)
  gemm_bf16_wmma<<<dim3(1, D_/64, B_), blk, 0, stream>>>(
      feature, (long)N_, 1L, (long)D_ * N_,
      score, 1L, (long)N_, (long)K_ * N_,
      mu, (long)K_, 1L, (long)D_ * K_, nullptr, N_);
  mu_div_k<<<(B_*D_*K_)/256, blk, 0, stream>>>(mu, pi);
  mu_norm_k<<<B_*K_, blk, 0, stream>>>(mu, invnmu);
  fnorm_k<<<(B_*N_)/256, blk, 0, stream>>>(feature, invnf);

  // fdot[n,k] = sum_d feature[d,n] * mu[d,k]   (A=feature^T via strides) -> cost1
  gemm_bf16_wmma<<<dim3(1, N_/64, B_), blk, 0, stream>>>(
      feature, 1L, (long)N_, (long)D_ * N_,
      mu, (long)K_, 1L, (long)D_ * K_,
      cost1, (long)K_, 1L, (long)N_ * K_, nullptr, D_);
  costfe_k<<<(B_*N_*K_)/256, blk, 0, stream>>>(cost1, invnf, invnmu);
  costeu_k<<<(B_*N_)/256, blk, 0, stream>>>(xyz, muxyz, yn, cost);

  // zero u, v, acc (contiguous region)
  int nz = 2*B_*N_ + 2*B_*K_ + 8;
  zero_f<<<(nz + 255)/256, blk, 0, stream>>>(uvec, nz);

  // 25 Sinkhorn iterations over 2 stacked problems
  for (int it = 0; it < 25; ++it) {
    sinku_k<<<(2*B_*N_)/256, blk, 0, stream>>>(cost, uvec, vvec);
    sinkv_k<<<2*B_*K_, blk, 0, stream>>>(cost, uvec, vvec);
  }

  regxyz_k<<<(B_*K_*K_)/256, blk, 0, stream>>>(muxyz, acc);
  regfea_k<<<B_*K_, blk, 0, stream>>>(mu, invnmu, acc);
  loss_k<<<(B_*N_)/256, blk, 0, stream>>>(logits, cost, uvec, vvec, acc);
  fin_k<<<1, 1, 0, stream>>>(acc, out);
}
